// MambaEncoder_17239998726191
// MI455X (gfx1250) — matrix operand order repaired and checked by reading, not compiled, not run on hardware
//
#include <hip/hip_runtime.h>
#include <math.h>

// Problem constants (from reference)
#define B_    2
#define L_    2048
#define DIN_  256
#define DM_   512
#define NL_   4
#define DS_   16
#define DC_   4
#define DI_   1024          // EXP * DM
#define DTR_  32            // (DM+15)//16
#define TOK_  (B_ * L_)     // 4096
#define DBLW_ (DTR_ + 2 * DS_)  // 64

typedef float  v2f  __attribute__((ext_vector_type(2)));
typedef float  v8f  __attribute__((ext_vector_type(8)));
typedef __bf16 v8bf  __attribute__((ext_vector_type(8)));
typedef __bf16 v16bf __attribute__((ext_vector_type(16)));

// ---------------------------------------------------------------------------
// bf16 fragment loaders (per-lane layouts used by this code, contiguous k order, 16-bit A/B)
// A 16x32 (MxK):  lane m=lane&15, half=lane>>4 holds A[m][half*16 + 0..15]
//                 (contiguous, same k order as B).
// B 32x16 (KxN):  lane n=lane&15, half=lane>>4 holds B[half*16 + 0..15][n],
//                 i.e. W[n][k + half*16 .. +15] for B = W^T (contiguous 32B).
// ---------------------------------------------------------------------------
__device__ __forceinline__ v16bf load_a_frag(const __bf16* p) {
  // p = row_base + k + half*16
  v8bf lo = *(const v8bf*)(p);        // 16B
  v8bf hi = *(const v8bf*)(p + 8);    // 16B
  v16bf r;
#pragma unroll
  for (int i = 0; i < 8; ++i) { r[i] = lo[i]; r[i + 8] = hi[i]; }
  return r;
}

// ---------------------------------------------------------------------------
// bf16 WMMA GEMM: D[M,N](f32) = epilogue( A[M,K] @ W[N,K]^T + bias )
// Wave computes 16(M) x 64(N); K-step 32; register double-buffered so next
// step's 10 x b128 loads overlap the current step's 4 WMMAs.
// ---------------------------------------------------------------------------
template <bool RESID>
__global__ __launch_bounds__(256) void gemm_wmma_bf16(
    const __bf16* __restrict__ A, int lda,
    const __bf16* __restrict__ W,          // N x K row-major bf16
    const float* __restrict__ bias,        // length N or nullptr
    float* __restrict__ D, int ldd,
    int M, int N, int K) {
  const int lane = threadIdx.x & 31;
  const int wave = threadIdx.x >> 5;
  const int tm   = (blockIdx.y * 8 + wave) * 16;
  const int tn0  = blockIdx.x * 64;
  if (tm >= M) return;

  const int mr   = lane & 15;
  const int half = lane >> 4;
  const int ncol = lane & 15;

  const __bf16* ap  = A + (size_t)(tm + mr) * lda + half * 16;
  const __bf16* wp0 = W + (size_t)(tn0 +  0 + ncol) * K + half * 16;
  const __bf16* wp1 = W + (size_t)(tn0 + 16 + ncol) * K + half * 16;
  const __bf16* wp2 = W + (size_t)(tn0 + 32 + ncol) * K + half * 16;
  const __bf16* wp3 = W + (size_t)(tn0 + 48 + ncol) * K + half * 16;

  v8f acc0 = {}, acc1 = {}, acc2 = {}, acc3 = {};

  // prologue: fragments for k = 0
  v16bf a_c  = load_a_frag(ap);
  v16bf b0_c = *(const v16bf*)(wp0);
  v16bf b1_c = *(const v16bf*)(wp1);
  v16bf b2_c = *(const v16bf*)(wp2);
  v16bf b3_c = *(const v16bf*)(wp3);

  for (int k = 32; k < K; k += 32) {
    // issue next step's loads first, then consume current regs -> loads
    // overlap the 4 multi-cycle WMMAs below.
    v16bf a_n  = load_a_frag(ap + k);
    v16bf b0_n = *(const v16bf*)(wp0 + k);
    v16bf b1_n = *(const v16bf*)(wp1 + k);
    v16bf b2_n = *(const v16bf*)(wp2 + k);
    v16bf b3_n = *(const v16bf*)(wp3 + k);

    acc0 = __builtin_amdgcn_wmma_f32_16x16x32_bf16(false, a_c, false, b0_c, (short)0, acc0, false, false);
    acc1 = __builtin_amdgcn_wmma_f32_16x16x32_bf16(false, a_c, false, b1_c, (short)0, acc1, false, false);
    acc2 = __builtin_amdgcn_wmma_f32_16x16x32_bf16(false, a_c, false, b2_c, (short)0, acc2, false, false);
    acc3 = __builtin_amdgcn_wmma_f32_16x16x32_bf16(false, a_c, false, b3_c, (short)0, acc3, false, false);

    a_c = a_n; b0_c = b0_n; b1_c = b1_n; b2_c = b2_n; b3_c = b3_n;
  }
  acc0 = __builtin_amdgcn_wmma_f32_16x16x32_bf16(false, a_c, false, b0_c, (short)0, acc0, false, false);
  acc1 = __builtin_amdgcn_wmma_f32_16x16x32_bf16(false, a_c, false, b1_c, (short)0, acc1, false, false);
  acc2 = __builtin_amdgcn_wmma_f32_16x16x32_bf16(false, a_c, false, b2_c, (short)0, acc2, false, false);
  acc3 = __builtin_amdgcn_wmma_f32_16x16x32_bf16(false, a_c, false, b3_c, (short)0, acc3, false, false);

  // C/D layout: VGPR r -> row (tm + r + 8*half), col = lane&15 within tile
#pragma unroll
  for (int t = 0; t < 4; ++t) {
    v8f acc = (t == 0) ? acc0 : (t == 1) ? acc1 : (t == 2) ? acc2 : acc3;
    const int n  = tn0 + t * 16 + ncol;
    const float bv = bias ? bias[n] : 0.0f;
#pragma unroll
    for (int r = 0; r < 8; ++r) {
      const int m = tm + r + 8 * half;
      float v = acc[r] + bv;
      float* dp = D + (size_t)m * ldd + n;
      if (RESID) v += *dp;
      *dp = v;
    }
  }
}

// ---------------------------------------------------------------------------
// f32 WMMA GEMM (V_WMMA_F32_16X16X4_F32) for the small, precision-sensitive
// GEMMs (x_proj -> dt_in|B|C, and dt with softplus epilogue).
// ACT: 0 = none, 2 = softplus.
// ---------------------------------------------------------------------------
template <int ACT>
__global__ __launch_bounds__(256) void gemm_wmma_f32(
    const float* __restrict__ A, int lda,
    const float* __restrict__ W,           // N x K row-major
    const float* __restrict__ bias,        // length N or nullptr
    float* __restrict__ D, int ldd,
    int M, int N, int K) {
  const int lane = threadIdx.x & 31;
  const int wave = threadIdx.x >> 5;
  const int tm   = (blockIdx.y * 8 + wave) * 16;
  const int tn0  = blockIdx.x * 64;
  if (tm >= M) return;

  const int mrow = tm + (lane & 15);
  const int kh   = (lane >> 4) * 2;
  const int ncol = lane & 15;

  const float* ap  = A + (size_t)mrow * lda + kh;
  const float* wp0 = W + (size_t)(tn0 +  0 + ncol) * K + kh;
  const float* wp1 = W + (size_t)(tn0 + 16 + ncol) * K + kh;
  const float* wp2 = W + (size_t)(tn0 + 32 + ncol) * K + kh;
  const float* wp3 = W + (size_t)(tn0 + 48 + ncol) * K + kh;

  v8f acc0 = {}, acc1 = {}, acc2 = {}, acc3 = {};
  v2f a_c  = *(const v2f*)(ap);
  v2f b0_c = *(const v2f*)(wp0);
  v2f b1_c = *(const v2f*)(wp1);
  v2f b2_c = *(const v2f*)(wp2);
  v2f b3_c = *(const v2f*)(wp3);
  for (int k = 4; k < K; k += 4) {
    v2f a_n  = *(const v2f*)(ap + k);
    v2f b0_n = *(const v2f*)(wp0 + k);
    v2f b1_n = *(const v2f*)(wp1 + k);
    v2f b2_n = *(const v2f*)(wp2 + k);
    v2f b3_n = *(const v2f*)(wp3 + k);
    acc0 = __builtin_amdgcn_wmma_f32_16x16x4_f32(false, a_c, false, b0_c, (short)0, acc0, false, false);
    acc1 = __builtin_amdgcn_wmma_f32_16x16x4_f32(false, a_c, false, b1_c, (short)0, acc1, false, false);
    acc2 = __builtin_amdgcn_wmma_f32_16x16x4_f32(false, a_c, false, b2_c, (short)0, acc2, false, false);
    acc3 = __builtin_amdgcn_wmma_f32_16x16x4_f32(false, a_c, false, b3_c, (short)0, acc3, false, false);
    a_c = a_n; b0_c = b0_n; b1_c = b1_n; b2_c = b2_n; b3_c = b3_n;
  }
  acc0 = __builtin_amdgcn_wmma_f32_16x16x4_f32(false, a_c, false, b0_c, (short)0, acc0, false, false);
  acc1 = __builtin_amdgcn_wmma_f32_16x16x4_f32(false, a_c, false, b1_c, (short)0, acc1, false, false);
  acc2 = __builtin_amdgcn_wmma_f32_16x16x4_f32(false, a_c, false, b2_c, (short)0, acc2, false, false);
  acc3 = __builtin_amdgcn_wmma_f32_16x16x4_f32(false, a_c, false, b3_c, (short)0, acc3, false, false);

  const int half = lane >> 4;
#pragma unroll
  for (int t = 0; t < 4; ++t) {
    v8f acc = (t == 0) ? acc0 : (t == 1) ? acc1 : (t == 2) ? acc2 : acc3;
    const int n  = tn0 + t * 16 + ncol;
    const float bv = bias ? bias[n] : 0.0f;
#pragma unroll
    for (int r = 0; r < 8; ++r) {
      const int m = tm + r + 8 * half;
      float v = acc[r] + bv;
      if (ACT == 2) v = (v > 20.0f) ? v : log1pf(expf(v));   // softplus
      D[(size_t)m * ldd + n] = v;
    }
  }
}

// ---------------------------------------------------------------------------
// f32 -> bf16 conversion (grid-stride)
// ---------------------------------------------------------------------------
__global__ __launch_bounds__(256) void cvt_bf16_kernel(
    const float* __restrict__ in, __bf16* __restrict__ out, int n) {
  for (int i = blockIdx.x * blockDim.x + threadIdx.x; i < n;
       i += gridDim.x * blockDim.x)
    out[i] = (__bf16)in[i];
}

// ---------------------------------------------------------------------------
// LayerNorm over last dim: one wave per token; output type templated so the
// per-layer LN feeds bf16 straight into the in_proj GEMM.
// ---------------------------------------------------------------------------
template <typename OT>
__global__ __launch_bounds__(256) void ln_kernel(
    const float* __restrict__ x, const float* __restrict__ w,
    const float* __restrict__ b, OT* __restrict__ o, int D) {
  const int lane = threadIdx.x & 31;
  const int wave = threadIdx.x >> 5;
  const int t    = blockIdx.x * 8 + wave;
  const float* xp = x + (size_t)t * D;
  float s = 0.0f, s2 = 0.0f;
  for (int i = lane; i < D; i += 32) {
    float v = xp[i];
    s += v; s2 += v * v;
  }
#pragma unroll
  for (int off = 16; off > 0; off >>= 1) {
    s  += __shfl_xor(s, off, 32);
    s2 += __shfl_xor(s2, off, 32);
  }
  const float mean = s / D;
  const float var  = s2 / D - mean * mean;
  const float inv  = rsqrtf(var + 1e-5f);
  OT* op = o + (size_t)t * D;
  for (int i = lane; i < D; i += 32)
    op[i] = (OT)((xp[i] - mean) * inv * w[i] + b[i]);
}

// ---------------------------------------------------------------------------
// Depthwise causal conv (DC=4 taps) + bias + SiLU.  Reads xc half of xz.
// ---------------------------------------------------------------------------
__global__ __launch_bounds__(256) void conv_silu_kernel(
    const float* __restrict__ xz, const float* __restrict__ cw,
    const float* __restrict__ cb, float* __restrict__ xc) {
  const int idx = blockIdx.x * blockDim.x + threadIdx.x;  // over TOK_*DI_
  const int d = idx % DI_;
  const int t = idx / DI_;       // token = b*L + l
  const int l = t % L_;
  float acc = cb[d];
#pragma unroll
  for (int j = 0; j < DC_; ++j) {
    const int ll = l - (DC_ - 1) + j;
    if (ll >= 0)
      acc += cw[d * DC_ + j] * xz[((size_t)t + (ll - l)) * (2 * DI_) + d];
  }
  xc[(size_t)idx] = acc / (1.0f + expf(-acc));  // silu
}

// ---------------------------------------------------------------------------
// Selective-scan: one thread per (b, d) channel; DS=16 state in registers.
// Fuses y = (scan + Dp*xc) * silu(z); emits bf16 (sole consumer: out GEMM).
// ---------------------------------------------------------------------------
__global__ __launch_bounds__(256) void scan_kernel(
    const float* __restrict__ dt, const float* __restrict__ xc,
    const float* __restrict__ dbl, const float* __restrict__ xz,
    const float* __restrict__ alog, const float* __restrict__ Dpar,
    __bf16* __restrict__ y) {
  const int g = blockIdx.x * blockDim.x + threadIdx.x;  // [0, B_*DI_)
  const int b = g / DI_;
  const int d = g % DI_;

  float Arow[DS_];
#pragma unroll
  for (int s = 0; s < DS_; ++s) Arow[s] = -expf(alog[d * DS_ + s]);
  float st[DS_];
#pragma unroll
  for (int s = 0; s < DS_; ++s) st[s] = 0.0f;
  const float dp = Dpar[d];

  for (int l = 0; l < L_; ++l) {
    const size_t tok = (size_t)b * L_ + l;
    const float dtv = dt[tok * DI_ + d];
    const float xv  = xc[tok * DI_ + d];
    const float zv  = xz[tok * (2 * DI_) + DI_ + d];
    const float* Bm = dbl + tok * DBLW_ + DTR_;        // wave-uniform broadcast
    const float* Cm = Bm + DS_;
    float accv = 0.0f;
#pragma unroll
    for (int s = 0; s < DS_; ++s) {
      const float da = expf(dtv * Arow[s]);
      st[s] = da * st[s] + (dtv * Bm[s]) * xv;
      accv += st[s] * Cm[s];
    }
    const float yv = accv + dp * xv;
    y[tok * DI_ + d] = (__bf16)(yv * (zv / (1.0f + expf(-zv))));
  }
}

// ---------------------------------------------------------------------------
extern "C" void kernel_launch(void* const* d_in, const int* in_sizes, int n_in,
                              void* d_out, int out_size, void* d_ws, size_t ws_size,
                              hipStream_t stream) {
  (void)in_sizes; (void)n_in; (void)out_size; (void)ws_size;
  const float* x      = (const float*)d_in[0];
  const float* proj_w = (const float*)d_in[1];
  const float* proj_b = (const float*)d_in[2];
  const float* ln_w   = (const float*)d_in[3];
  const float* ln_b   = (const float*)d_in[4];
  const float* ipw    = (const float*)d_in[5];
  const float* cw     = (const float*)d_in[6];
  const float* cb     = (const float*)d_in[7];
  const float* xpw    = (const float*)d_in[8];
  const float* dtw    = (const float*)d_in[9];
  const float* dtb    = (const float*)d_in[10];
  const float* alog   = (const float*)d_in[11];
  const float* Dpar   = (const float*)d_in[12];
  const float* ow     = (const float*)d_in[13];
  const float* fnw    = (const float*)d_in[14];
  const float* fnb    = (const float*)d_in[15];

  // ---- workspace carve-up --------------------------------------------------
  float* ws  = (float*)d_ws;
  size_t off = 0;
  float* h   = ws + off; off += (size_t)TOK_ * DM_;         // residual stream
  float* xz  = ws + off; off += (size_t)TOK_ * 2 * DI_;     // in_proj output
  float* xc  = ws + off; off += (size_t)TOK_ * DI_;         // conv+silu (f32)
  float* dbl = ws + off; off += (size_t)TOK_ * DBLW_;       // [dt_in|B|C]
  float* dt  = ws + off; off += (size_t)TOK_ * DI_;         // softplus dt
  __bf16* bws = (__bf16*)(ws + off);
  size_t bo = 0;
  __bf16* xb   = bws + bo; bo += (size_t)TOK_ * DIN_;       // x in bf16
  __bf16* pwb  = bws + bo; bo += (size_t)DM_ * DIN_;        // proj_w bf16
  __bf16* ipwb = bws + bo; bo += (size_t)NL_ * 2 * DI_ * DM_;
  __bf16* owb  = bws + bo; bo += (size_t)NL_ * DM_ * DI_;
  __bf16* hnb  = bws + bo; bo += (size_t)TOK_ * DM_;        // LN output bf16
  __bf16* yb   = bws + bo;                                  // scan output bf16

  const dim3 blk(256);

  // ---- one-time bf16 conversions (weights + x) -----------------------------
  cvt_bf16_kernel<<<1024, blk, 0, stream>>>(x, xb, TOK_ * DIN_);
  cvt_bf16_kernel<<<512,  blk, 0, stream>>>(proj_w, pwb, DM_ * DIN_);
  cvt_bf16_kernel<<<2048, blk, 0, stream>>>(ipw, ipwb, NL_ * 2 * DI_ * DM_);
  cvt_bf16_kernel<<<2048, blk, 0, stream>>>(ow, owb, NL_ * DM_ * DI_);

  // h = x @ proj_w^T + proj_b           (M=4096, N=512, K=256)
  gemm_wmma_bf16<false><<<dim3(DM_ / 64, TOK_ / 128), blk, 0, stream>>>(
      xb, DIN_, pwb, proj_b, h, DM_, TOK_, DM_, DIN_);

  for (int l = 0; l < NL_; ++l) {
    // hnb = LN(h) in bf16
    ln_kernel<__bf16><<<TOK_ / 8, blk, 0, stream>>>(
        h, ln_w + l * DM_, ln_b + l * DM_, hnb, DM_);
    // xz = hn @ in_proj^T                (N=2048, K=512)
    gemm_wmma_bf16<false><<<dim3(2 * DI_ / 64, TOK_ / 128), blk, 0, stream>>>(
        hnb, DM_, ipwb + (size_t)l * 2 * DI_ * DM_, nullptr, xz, 2 * DI_, TOK_, 2 * DI_, DM_);
    // xc = silu(depthwise_conv(xz[:, :DI]) + conv_b)
    conv_silu_kernel<<<(TOK_ * DI_) / 256, blk, 0, stream>>>(
        xz, cw + l * DI_ * DC_, cb + l * DI_, xc);
    // dbl = xc @ xproj^T                 (N=64, K=1024)  f32 WMMA (precision)
    gemm_wmma_f32<0><<<dim3(DBLW_ / 64, TOK_ / 128), blk, 0, stream>>>(
        xc, DI_, xpw + (size_t)l * DBLW_ * DI_, nullptr, dbl, DBLW_, TOK_, DBLW_, DI_);
    // dt = softplus(dbl[:, :32] @ dt_w^T + dt_b)   (N=1024, K=32)  f32 WMMA
    gemm_wmma_f32<2><<<dim3(DI_ / 64, TOK_ / 128), blk, 0, stream>>>(
        dbl, DBLW_, dtw + (size_t)l * DI_ * DTR_, dtb + l * DI_, dt, DI_, TOK_, DI_, DTR_);
    // y = scan(dt, xc, Bm, Cm) fused with Dp*xc and silu(z) gate -> bf16
    scan_kernel<<<(B_ * DI_) / 256, blk, 0, stream>>>(
        dt, xc, dbl, xz, alog + (size_t)l * DI_ * DS_, Dpar + l * DI_, yb);
    // h += y @ out_w^T                   (N=512, K=1024), residual
    gemm_wmma_bf16<true><<<dim3(DM_ / 64, TOK_ / 128), blk, 0, stream>>>(
        yb, DI_, owb + (size_t)l * DM_ * DI_, nullptr, h, DM_, TOK_, DM_, DI_);
  }
  // out = LN(h) in f32
  ln_kernel<float><<<TOK_ / 8, blk, 0, stream>>>(h, fnw, fnb, (float*)d_out, DM_);
}
